// GNN_55808805044485
// MI455X (gfx1250) — matrix operand order repaired and checked by reading, not compiled, not run on hardware
//
#include <hip/hip_runtime.h>
#include <hip/hip_bf16.h>

typedef __attribute__((ext_vector_type(16))) __bf16 v16bf;
typedef __attribute__((ext_vector_type(8)))  float  v8f;

union BF16x16 {
    unsigned short u[16];
    uint4          q[2];
    v16bf          v;
};

__device__ __forceinline__ unsigned short f2bf(float f) {
    unsigned int u = __float_as_uint(f);
    unsigned int r = u + 0x7FFFu + ((u >> 16) & 1u);   // round-to-nearest-even
    return (unsigned short)(r >> 16);
}

// ---------------------------------------------------------------------------
// Pack W[K,F] (fp32 row-major) into bf16 WMMA B-fragments.
// Fragment (kblk, fblk): lane n (0-15) holds col n, K octets {0..7, 16..23};
// lane n+16 holds col n, K octets {8..15, 24..31}. Stored [frag][lane][16] bf16.
// ---------------------------------------------------------------------------
__global__ void pack_w_kernel(const float* __restrict__ W,
                              unsigned short* __restrict__ Wp,
                              int K, int F) {
    int t = blockIdx.x * blockDim.x + threadIdx.x;
    int fblocks = F >> 4;
    int total = (K >> 5) * fblocks * 32;
    if (t >= total) return;
    int lane = t & 31;
    int frag = t >> 5;
    int kblk = frag / fblocks;
    int fblk = frag % fblocks;
    int col  = fblk * 16 + (lane & 15);
    int krow = kblk * 32 + ((lane >> 4) << 3);
    unsigned short* dst = Wp + (size_t)frag * 512 + lane * 16;
#pragma unroll
    for (int i = 0; i < 16; ++i)
        dst[i] = f2bf(W[(size_t)(krow + (i & 7) + ((i >> 3) << 4)) * F + col]);
}

// ---------------------------------------------------------------------------
// Pack activations X[M,K] (fp32) into bf16 WMMA A-fragments, optional ReLU.
// A-frag (16x32, 16-bit): lane L<16 holds row M=L, K octets {0..7, 16..23};
// lanes 16-31 hold octets {8..15, 24..31}. Stored [mt*kblocks+kb][lane][16].
// ---------------------------------------------------------------------------
__global__ void pack_a_kernel(const float* __restrict__ X,
                              unsigned short* __restrict__ Ap,
                              int K, long long total, int doRelu) {
    long long t = (long long)blockIdx.x * blockDim.x + threadIdx.x;
    if (t >= total) return;
    int lane = (int)(t & 31);
    long long frag = t >> 5;
    int kblocks = K >> 5;
    int mt = (int)(frag / kblocks);
    int kb = (int)(frag % kblocks);
    int row   = (mt << 4) + (lane & 15);
    int kbase = (kb << 5) + ((lane >> 4) << 3);
    const float* xr = X + (size_t)row * K + kbase;
    unsigned short* dst = Ap + (size_t)frag * 512 + lane * 16;
#pragma unroll
    for (int h = 0; h < 2; ++h) {           // K octet pair: +0 and +16
        float4 a = *(const float4*)(xr + h * 16);
        float4 b = *(const float4*)(xr + h * 16 + 4);
        if (doRelu) {
            a.x = fmaxf(a.x, 0.f); a.y = fmaxf(a.y, 0.f);
            a.z = fmaxf(a.z, 0.f); a.w = fmaxf(a.w, 0.f);
            b.x = fmaxf(b.x, 0.f); b.y = fmaxf(b.y, 0.f);
            b.z = fmaxf(b.z, 0.f); b.w = fmaxf(b.w, 0.f);
        }
        dst[h * 8 + 0] = f2bf(a.x); dst[h * 8 + 1] = f2bf(a.y);
        dst[h * 8 + 2] = f2bf(a.z); dst[h * 8 + 3] = f2bf(a.w);
        dst[h * 8 + 4] = f2bf(b.x); dst[h * 8 + 5] = f2bf(b.y);
        dst[h * 8 + 6] = f2bf(b.z); dst[h * 8 + 7] = f2bf(b.w);
    }
}

// ---------------------------------------------------------------------------
// H[M,F] = A @ W from pre-packed bf16 fragments. One wave = 16x128 tile
// (8 accumulators); inner loop is pure b128 loads + 8 WMMAs per K-step.
// ---------------------------------------------------------------------------
__global__ __launch_bounds__(256)
void gemm_wmma_kernel(const unsigned short* __restrict__ Ap,
                      const unsigned short* __restrict__ Wp,
                      float* __restrict__ H,
                      int M, int K, int F) {
    const int lane   = threadIdx.x & 31;
    const int ntiles = F >> 7;               // 128-wide N tiles (F=256 -> 2)
    const int mtiles = M >> 4;               // 50000/16 = 3125 exact
    int waveId = blockIdx.x * (blockDim.x >> 5) + (threadIdx.x >> 5);
    if (waveId >= mtiles * ntiles) return;   // wave-uniform: EXEC stays all-1s

    const int mt = waveId / ntiles;
    const int nt = waveId % ntiles;
    const int f0 = nt << 7;
    const int kblocks = K >> 5;
    const int fblocks = F >> 4;

    v8f c[8] = {{}, {}, {}, {}, {}, {}, {}, {}};

    // A fragments: [mt*kblocks + kb][lane][16]; step = 32 uint4 = 256 ush
    const uint4* ap = (const uint4*)Ap + ((size_t)mt * kblocks) * 32 + lane * 2;
    // B fragments: [kb*fblocks + (f0>>4)+t][lane][16]; kb step = fblocks*32 uint4
    const uint4* wp = (const uint4*)Wp + (size_t)(f0 >> 4) * 32 + lane * 2;
    const int wstep = fblocks * 32;

    for (int kb = 0; kb < kblocks; ++kb) {
        BF16x16 a;
        a.q[0] = ap[0]; a.q[1] = ap[1];
        ap += 32;

        BF16x16 b0, b1, b2, b3;
        b0.q[0] = wp[0];  b0.q[1] = wp[1];
        b1.q[0] = wp[32]; b1.q[1] = wp[33];
        b2.q[0] = wp[64]; b2.q[1] = wp[65];
        b3.q[0] = wp[96]; b3.q[1] = wp[97];
        c[0] = __builtin_amdgcn_wmma_f32_16x16x32_bf16(false, a.v, false, b0.v,
                                                       (short)0, c[0], false, false);
        c[1] = __builtin_amdgcn_wmma_f32_16x16x32_bf16(false, a.v, false, b1.v,
                                                       (short)0, c[1], false, false);
        c[2] = __builtin_amdgcn_wmma_f32_16x16x32_bf16(false, a.v, false, b2.v,
                                                       (short)0, c[2], false, false);
        c[3] = __builtin_amdgcn_wmma_f32_16x16x32_bf16(false, a.v, false, b3.v,
                                                       (short)0, c[3], false, false);

        b0.q[0] = wp[128]; b0.q[1] = wp[129];
        b1.q[0] = wp[160]; b1.q[1] = wp[161];
        b2.q[0] = wp[192]; b2.q[1] = wp[193];
        b3.q[0] = wp[224]; b3.q[1] = wp[225];
        wp += wstep;
        c[4] = __builtin_amdgcn_wmma_f32_16x16x32_bf16(false, a.v, false, b0.v,
                                                       (short)0, c[4], false, false);
        c[5] = __builtin_amdgcn_wmma_f32_16x16x32_bf16(false, a.v, false, b1.v,
                                                       (short)0, c[5], false, false);
        c[6] = __builtin_amdgcn_wmma_f32_16x16x32_bf16(false, a.v, false, b2.v,
                                                       (short)0, c[6], false, false);
        c[7] = __builtin_amdgcn_wmma_f32_16x16x32_bf16(false, a.v, false, b3.v,
                                                       (short)0, c[7], false, false);
    }

    // C/D layout: VGPR v, lanes 0-15 -> M=m0+v, N=f0+lane; lanes 16-31 -> M+8
    const int cr = (mt << 4) + ((lane >> 4) << 3);
    const int cc = f0 + (lane & 15);
    float* hp = H + (size_t)cr * F + cc;
#pragma unroll
    for (int v = 0; v < 8; ++v) {
#pragma unroll
        for (int t = 0; t < 8; ++t)
            hp[(size_t)v * F + t * 16] = c[t][v];
    }
}

// agg[i, :] = b[:]
__global__ void init_bias_kernel(float* __restrict__ agg,
                                 const float* __restrict__ b,
                                 long long total) {
    long long g = (long long)blockIdx.x * blockDim.x + threadIdx.x;
    if (g < total) agg[g] = b[g & 255];
}

// agg[dst[e], :] += h[src[e], :]   (32 threads/edge, 8 floats each)
__global__ void scatter_add_kernel(const float* __restrict__ H,
                                   const int* __restrict__ src,
                                   const int* __restrict__ dst,
                                   float* __restrict__ agg,
                                   long long total /* = E*32 */) {
    long long g = (long long)blockIdx.x * blockDim.x + threadIdx.x;
    if (g >= total) return;
    int e = (int)(g >> 5);
    int q = (int)(g & 31);
    int s = src[e];
    int d = dst[e];
    const float4* hv = (const float4*)(H + (size_t)s * 256 + q * 8);
    float4 v0 = hv[0], v1 = hv[1];
    float* base = agg + (size_t)d * 256 + q * 8;
    atomicAdd(base + 0, v0.x); atomicAdd(base + 1, v0.y);
    atomicAdd(base + 2, v0.z); atomicAdd(base + 3, v0.w);
    atomicAdd(base + 4, v1.x); atomicAdd(base + 5, v1.y);
    atomicAdd(base + 6, v1.z); atomicAdd(base + 7, v1.w);
}

__global__ void relu_kernel(const float* __restrict__ in,
                            float* __restrict__ out, long long total) {
    long long g = (long long)blockIdx.x * blockDim.x + threadIdx.x;
    if (g < total) out[g] = fmaxf(in[g], 0.0f);
}

// ---------------------------------------------------------------------------
extern "C" void kernel_launch(void* const* d_in, const int* in_sizes, int n_in,
                              void* d_out, int out_size, void* d_ws, size_t ws_size,
                              hipStream_t stream) {
    const float* x    = (const float*)d_in[0];
    const int*   edge = (const int*)  d_in[1];
    const float* W1   = (const float*)d_in[2];
    const float* b1   = (const float*)d_in[3];
    const float* W2   = (const float*)d_in[4];
    const float* b2   = (const float*)d_in[5];
    const float* W3   = (const float*)d_in[6];
    const float* b3   = (const float*)d_in[7];
    float* out = (float*)d_out;

    const int N = in_sizes[0] / 128;      // 50000
    const int E = in_sizes[1] / 2;        // 800000
    const int F = 256;
    const int K1 = 128;

    const int* src = edge;
    const int* dst = edge + E;

    const size_t szNF = (size_t)N * F * sizeof(float);          // 51.2 MB
    char* w = (char*)d_ws;
    float* A            = (float*)w;                            // h (GEMM out)
    float* B            = (float*)(w + szNF);                   // agg buffer
    unsigned short* Ap  = (unsigned short*)(w + 2 * szNF);      // packed bf16 acts
    unsigned short* Wp1 = Ap  + (size_t)N * F;                  // N*256 bf16
    unsigned short* Wp2 = Wp1 + (size_t)K1 * F;
    unsigned short* Wp3 = Wp2 + (size_t)F  * F;

    const long long totNF = (long long)N * F;
    const long long totE  = (long long)E * 32;
    const int TB = 256;
    const int gemmTiles  = (N / 16) * (F / 128);                // 6250
    const int gemmBlocks = (gemmTiles + 7) / 8;
    const int nfBlocks   = (int)((totNF + TB - 1) / TB);
    const int scBlocks   = (int)((totE + TB - 1) / TB);
    const long long pa1  = (long long)(N / 16) * (K1 / 32) * 32; // pack-A threads
    const long long pa2  = (long long)(N / 16) * (F / 32) * 32;
    const int pa1Blocks  = (int)((pa1 + TB - 1) / TB);
    const int pa2Blocks  = (int)((pa2 + TB - 1) / TB);

    // ---- pack weights (tiny, reused 3125x) ----
    {
        int t1 = (K1 / 32) * (F / 16) * 32;
        int t2 = (F  / 32) * (F / 16) * 32;
        pack_w_kernel<<<(t1 + TB - 1) / TB, TB, 0, stream>>>(W1, Wp1, K1, F);
        pack_w_kernel<<<(t2 + TB - 1) / TB, TB, 0, stream>>>(W2, Wp2, F,  F);
        pack_w_kernel<<<(t2 + TB - 1) / TB, TB, 0, stream>>>(W3, Wp3, F,  F);
    }

    // ---- Layer 1 ----
    pack_a_kernel<<<pa1Blocks, TB, 0, stream>>>(x, Ap, K1, pa1, 0);
    gemm_wmma_kernel<<<gemmBlocks, TB, 0, stream>>>(Ap, Wp1, A, N, K1, F);
    init_bias_kernel<<<nfBlocks, TB, 0, stream>>>(B, b1, totNF);
    scatter_add_kernel<<<scBlocks, TB, 0, stream>>>(A, src, dst, B, totE);

    // ---- Layer 2 (ReLU fused into A-packing) ----
    pack_a_kernel<<<pa2Blocks, TB, 0, stream>>>(B, Ap, F, pa2, 1);
    gemm_wmma_kernel<<<gemmBlocks, TB, 0, stream>>>(Ap, Wp2, A, N, F, F);
    init_bias_kernel<<<nfBlocks, TB, 0, stream>>>(B, b2, totNF);
    scatter_add_kernel<<<scBlocks, TB, 0, stream>>>(A, src, dst, B, totE);

    // ---- Layer 3 ----
    pack_a_kernel<<<pa2Blocks, TB, 0, stream>>>(B, Ap, F, pa2, 1);
    gemm_wmma_kernel<<<gemmBlocks, TB, 0, stream>>>(Ap, Wp3, A, N, F, F);
    init_bias_kernel<<<nfBlocks, TB, 0, stream>>>(B, b3, totNF);
    scatter_add_kernel<<<scBlocks, TB, 0, stream>>>(A, src, dst, B, totE);

    // ---- final ReLU -> d_out ----
    relu_kernel<<<nfBlocks, TB, 0, stream>>>(B, out, totNF);
}
